// LorentzResBlock_89635967468237
// MI455X (gfx1250) — matrix operand order repaired and checked by reading, not compiled, not run on hardware
//
#include <hip/hip_runtime.h>
#include <hip/hip_bf16.h>
#include <math.h>

// ---------- types ----------
typedef __attribute__((ext_vector_type(16))) __bf16 v16bf;
typedef __attribute__((ext_vector_type(8)))  float  v8f;
typedef __attribute__((ext_vector_type(4)))  float  v4f;

// ---------- problem constants ----------
#define BB   16
#define HH   128
#define WW   128
#define NPIX (BB*HH*WW)          // 262144
#define MPAD 64                  // 63 real output channels, padded to 64
#define SO   63

// ---------- pack x: NCHW f32 -> NHWC bf16 (C=32) ----------
__global__ __launch_bounds__(256) void pack_x_kernel(const float* __restrict__ x,
                                                     __bf16* __restrict__ xb) {
    long i = (long)blockIdx.x * 256 + threadIdx.x;        // over NPIX*32
    if (i >= (long)NPIX * 32) return;
    int c = (int)(i & 31);
    long p = i >> 5;                                      // pixel index (b*H+y)*W+x
    long b = p >> 14;                                     // / (128*128)
    long rem = p & 16383;
    xb[i] = (__bf16)x[((b * 32 + c) << 14) + rem];
}

// ---------- pack weights: OIHW f32 -> fragment-ordered [64][K] bf16 ----------
// Within each 32-wide K block, half h (lanes 16h..16h+15 of a wave) must hold
// K_local = {16h..16h+15} (contiguous, matching the B operand), so a lane
// can fetch its whole fragment with one contiguous 32-byte load.
__global__ __launch_bounds__(256) void pack_w_kernel(const float* __restrict__ w,
                                                     __bf16* __restrict__ A,
                                                     int C, int KH) {
    int K = KH * KH * C;
    long i = (long)blockIdx.x * 256 + threadIdx.x;        // over 64*K
    if (i >= (long)MPAD * K) return;
    int m = (int)(i / K);
    int r = (int)(i % K);
    int kblk   = r >> 5;
    int within = r & 31;
    int h = within >> 4;          // which half-wave consumes this element
    int j = within & 15;          // position within the lane's 16-element fragment
    int k_local = 16 * h + j;
    int k = kblk * 32 + k_local;  // logical K index: (ky*KH+kx)*C + c
    int kk = k / C, c = k % C;
    int ky = kk / KH, kx = kk % KH;
    __bf16 v = (__bf16)0.0f;
    if (m < SO) v = (__bf16)w[((m * C + c) * KH + ky) * KH + kx];
    A[i] = v;
}

// ---------- implicit-GEMM conv via WMMA bf16 ----------
// act: NHWC bf16, CIN channels. A: fragment-ordered [64][Kdim] bf16. outS: NHWC64 f32.
// Workgroup = 256 threads = 8 waves; wave w handles pixels x in [16w, 16w+16) of one row.
template <int CIN, int KH>
__global__ __launch_bounds__(256) void conv_wmma_kernel(const __bf16* __restrict__ act,
                                                        const __bf16* __restrict__ A,
                                                        const float*  __restrict__ bias,
                                                        float* __restrict__ outS) {
    constexpr int Kdim = KH * KH * CIN;
    constexpr int PAD  = (KH == 3) ? 1 : 0;
    constexpr int KSTEPS = Kdim / 32;

    const int bid  = blockIdx.x;          // b*H + y
    const int b    = bid >> 7;
    const int y    = bid & 127;
    const int wave = threadIdx.x >> 5;
    const int lane = threadIdx.x & 31;
    const int nCol = lane & 15;           // N column within 16-pixel strip
    const int khalf = lane >> 4;          // half-wave: selects A/B K sub-ranges
    const int x    = wave * 16 + nCol;

    v8f acc[4];
    acc[0] = v8f{}; acc[1] = v8f{}; acc[2] = v8f{}; acc[3] = v8f{};

    const int arow = lane & 15;
    const __bf16* a0 = A + (long)(arow +  0) * Kdim;
    const __bf16* a1 = A + (long)(arow + 16) * Kdim;
    const __bf16* a2 = A + (long)(arow + 32) * Kdim;
    const __bf16* a3 = A + (long)(arow + 48) * Kdim;

    for (int kk = 0; kk < KSTEPS; ++kk) {
        const int flat = kk * 32;
        const int ky = flat / (KH * CIN);
        const int kx = (flat / CIN) % KH;
        const int c0 = flat % CIN;
        const int koff = flat + khalf * 16;   // fragment-ordered A: contiguous 32B

        v16bf fa0 = *(const v16bf*)(a0 + koff);
        v16bf fa1 = *(const v16bf*)(a1 + koff);
        v16bf fa2 = *(const v16bf*)(a2 + koff);
        v16bf fa3 = *(const v16bf*)(a3 + koff);

        const int iy = y + ky - PAD;
        const int ix = x + kx - PAD;
        v16bf fb = {};
        if ((unsigned)iy < (unsigned)HH && (unsigned)ix < (unsigned)WW) {
            const long pix = ((long)b * HH + iy) * WW + ix;
            fb = *(const v16bf*)(act + pix * CIN + c0 + khalf * 16);
        }

        acc[0] = __builtin_amdgcn_wmma_f32_16x16x32_bf16(false, fa0, false, fb, (short)0, acc[0], false, false);
        acc[1] = __builtin_amdgcn_wmma_f32_16x16x32_bf16(false, fa1, false, fb, (short)0, acc[1], false, false);
        acc[2] = __builtin_amdgcn_wmma_f32_16x16x32_bf16(false, fa2, false, fb, (short)0, acc[2], false, false);
        acc[3] = __builtin_amdgcn_wmma_f32_16x16x32_bf16(false, fa3, false, fb, (short)0, acc[3], false, false);
    }

    // Epilogue: C layout -> lane holds (r, lane): m = 16t + 8*khalf + r, n = lane&15
    const long pix = ((long)b * HH + y) * WW + x;
    float* op = outS + pix * MPAD;
#pragma unroll
    for (int t = 0; t < 4; ++t) {
        const int m0 = t * 16 + khalf * 8;
        v4f lo, hi;
#pragma unroll
        for (int r = 0; r < 4; ++r) {
            const int m = m0 + r;
            lo[r] = acc[t][r] + ((m < SO) ? bias[m] : 0.0f);
        }
#pragma unroll
        for (int r = 4; r < 8; ++r) {
            const int m = m0 + r;
            hi[r - 4] = acc[t][r] + ((m < SO) ? bias[m] : 0.0f);
        }
        *(v4f*)(op + m0)     = lo;
        *(v4f*)(op + m0 + 4) = hi;
    }
}

// ---------- per-channel partial sums (deterministic two-stage BN reduction) ----------
#define STAT_BLOCKS 1024
__global__ __launch_bounds__(256) void stats_kernel(const float* __restrict__ s,
                                                    float* __restrict__ part) {
    __shared__ float sh[512];
    const int c   = threadIdx.x & 63;
    const int sub = threadIdx.x >> 6;     // 0..3
    float sum = 0.f, sq = 0.f;
    for (int p = blockIdx.x * 4 + sub; p < NPIX; p += STAT_BLOCKS * 4) {
        const float v = s[(long)p * MPAD + c];
        sum += v; sq += v * v;
    }
    sh[threadIdx.x]       = sum;
    sh[256 + threadIdx.x] = sq;
    __syncthreads();
    if (sub == 0) {
        const float S = sh[c] + sh[c + 64] + sh[c + 128] + sh[c + 192];
        const float Q = sh[256 + c] + sh[256 + c + 64] + sh[256 + c + 128] + sh[256 + c + 192];
        part[blockIdx.x * 128 + c]      = S;
        part[blockIdx.x * 128 + 64 + c] = Q;
    }
}

__global__ void bn_finalize_kernel(const float* __restrict__ part,
                                   const float* __restrict__ gamma,
                                   const float* __restrict__ beta,
                                   float* __restrict__ scale,
                                   float* __restrict__ shift) {
    const int c = threadIdx.x;            // 64 threads
    float S = 0.f, Q = 0.f;
    for (int b = 0; b < STAT_BLOCKS; ++b) {
        S += part[b * 128 + c];
        Q += part[b * 128 + 64 + c];
    }
    if (c < SO) {
        const float invN = 1.0f / (float)NPIX;
        const float mu  = S * invN;
        const float var = Q * invN - mu * mu;
        const float sc  = gamma[c] * rsqrtf(var + 1e-5f);
        scale[c] = sc;
        shift[c] = beta[c] - mu * sc;
    } else {
        scale[c] = 0.f;                   // padded channel contributes nothing
        shift[c] = 0.f;
    }
}

// ---------- BN + ReLU + Lorentz projection -> bf16 activation (conv2 input) ----------
// Wave-per-pixel: lane c handles channels c and c+32 (coalesced), wave32
// shuffle reduction produces the time channel t = sqrt(K + |space|^2).
__global__ __launch_bounds__(256) void act_kernel(const float* __restrict__ s1,
                                                  const float* __restrict__ scale,
                                                  const float* __restrict__ shift,
                                                  __bf16* __restrict__ act) {
    const int wave = threadIdx.x >> 5;
    const int lane = threadIdx.x & 31;
    const int p = blockIdx.x * 8 + wave;
    if (p >= NPIX) return;
    const float* sp = s1 + (long)p * MPAD;
    __bf16* ap = act + (long)p * MPAD;

    const int c0 = lane, c1 = lane + 32;
    float v0 = fmaxf(sp[c0] * scale[c0] + shift[c0], 0.f);
    float v1 = fmaxf(sp[c1] * scale[c1] + shift[c1], 0.f);   // c1==63 -> 0 (scale/shift zeroed)

    float ss = v0 * v0 + v1 * v1;
#pragma unroll
    for (int off = 16; off > 0; off >>= 1)
        ss += __shfl_xor(ss, off, 32);

    ap[c0 + 1] = (__bf16)v0;
    if (c1 < SO) ap[c1 + 1] = (__bf16)v1;
    if (lane == 0) ap[0] = (__bf16)sqrtf(1.0f + ss);
}

// ---------- final: BN2 + BNp + residual + ReLU + projection -> NCHW f32 out ----------
__global__ __launch_bounds__(256) void final_kernel(const float* __restrict__ s2,
                                                    const float* __restrict__ spw,
                                                    const float* __restrict__ scale2,
                                                    const float* __restrict__ shift2,
                                                    const float* __restrict__ scalep,
                                                    const float* __restrict__ shiftp,
                                                    float* __restrict__ out) {
    const int p = blockIdx.x * 256 + threadIdx.x;
    if (p >= NPIX) return;
    const int b   = p >> 14;
    const int rem = p & 16383;
    float* op = out + (long)b * 64 * 16384 + rem;
    const float* a = s2  + (long)p * MPAD;
    const float* q = spw + (long)p * MPAD;
    float ss = 0.f;
    for (int c = 0; c < SO; ++c) {
        float v = a[c] * scale2[c] + shift2[c] + q[c] * scalep[c] + shiftp[c];
        v = fmaxf(v, 0.f);
        op[(long)(c + 1) * 16384] = v;
        ss += v * v;
    }
    op[0] = sqrtf(1.0f + ss);
}

// ---------- host side ----------
extern "C" void kernel_launch(void* const* d_in, const int* in_sizes, int n_in,
                              void* d_out, int out_size, void* d_ws, size_t ws_size,
                              hipStream_t stream) {
    const float* x   = (const float*)d_in[0];
    const float* w1  = (const float*)d_in[1];
    const float* b1  = (const float*)d_in[2];
    const float* g1  = (const float*)d_in[3];
    const float* bt1 = (const float*)d_in[4];
    const float* w2  = (const float*)d_in[5];
    const float* b2  = (const float*)d_in[6];
    const float* g2  = (const float*)d_in[7];
    const float* bt2 = (const float*)d_in[8];
    const float* wp  = (const float*)d_in[9];
    const float* bp  = (const float*)d_in[10];
    const float* gp  = (const float*)d_in[11];
    const float* btp = (const float*)d_in[12];
    float* out = (float*)d_out;

    // Workspace layout (bytes, all 256-aligned)
    char* ws = (char*)d_ws;
    const size_t OFF_XB   = 0;                            // NPIX*32 bf16 = 16 MiB
    const size_t OFF_A1   = OFF_XB  + (size_t)NPIX * 32 * 2;       // 64*288 bf16
    const size_t OFF_A2   = OFF_A1  + (size_t)MPAD * 288 * 2;      // 64*576 bf16
    const size_t OFF_AP   = OFF_A2  + (size_t)MPAD * 576 * 2;      // 64*32 bf16
    const size_t OFF_S1   = OFF_AP  + (size_t)MPAD * 32 * 2;       // NPIX*64 f32 (also s2)
    const size_t OFF_ACT  = OFF_S1  + (size_t)NPIX * MPAD * 4;     // NPIX*64 bf16
    const size_t OFF_SP   = OFF_ACT + (size_t)NPIX * MPAD * 2;     // NPIX*64 f32
    const size_t OFF_PART = OFF_SP  + (size_t)NPIX * MPAD * 4;     // 1024*128 f32
    const size_t OFF_SS   = OFF_PART + (size_t)STAT_BLOCKS * 128 * 4; // 6*64 f32

    __bf16* xb   = (__bf16*)(ws + OFF_XB);
    __bf16* A1   = (__bf16*)(ws + OFF_A1);
    __bf16* A2   = (__bf16*)(ws + OFF_A2);
    __bf16* AP   = (__bf16*)(ws + OFF_AP);
    float*  s1   = (float*)(ws + OFF_S1);       // reused as s2 after act pack
    __bf16* act  = (__bf16*)(ws + OFF_ACT);
    float*  spw  = (float*)(ws + OFF_SP);
    float*  part = (float*)(ws + OFF_PART);
    float*  scale1 = (float*)(ws + OFF_SS);
    float*  shift1 = scale1 + 64;
    float*  scale2 = scale1 + 128;
    float*  shift2 = scale1 + 192;
    float*  scalep = scale1 + 256;
    float*  shiftp = scale1 + 320;

    // 1) pack inputs to bf16 (A matrices in WMMA fragment order)
    pack_x_kernel<<<((long)NPIX * 32 + 255) / 256, 256, 0, stream>>>(x, xb);
    pack_w_kernel<<<(MPAD * 288 + 255) / 256, 256, 0, stream>>>(w1, A1, 32, 3);
    pack_w_kernel<<<(MPAD * 576 + 255) / 256, 256, 0, stream>>>(w2, A2, 64, 3);
    pack_w_kernel<<<(MPAD * 32  + 255) / 256, 256, 0, stream>>>(wp, AP, 32, 1);

    // 2) conv1 (3x3, Cin=32) -> s1 ; BN1 stats ; activation pack
    conv_wmma_kernel<32, 3><<<BB * HH, 256, 0, stream>>>(xb, A1, b1, s1);
    stats_kernel<<<STAT_BLOCKS, 256, 0, stream>>>(s1, part);
    bn_finalize_kernel<<<1, 64, 0, stream>>>(part, g1, bt1, scale1, shift1);
    act_kernel<<<(NPIX + 7) / 8, 256, 0, stream>>>(s1, scale1, shift1, act);

    // 3) conv2 (3x3, Cin=64) -> s2 (aliases s1) ; BN2 stats
    conv_wmma_kernel<64, 3><<<BB * HH, 256, 0, stream>>>(act, A2, b2, s1);
    stats_kernel<<<STAT_BLOCKS, 256, 0, stream>>>(s1, part);
    bn_finalize_kernel<<<1, 64, 0, stream>>>(part, g2, bt2, scale2, shift2);

    // 4) pointwise shortcut conv (1x1, Cin=32) -> spw ; BNp stats
    conv_wmma_kernel<32, 1><<<BB * HH, 256, 0, stream>>>(xb, AP, bp, spw);
    stats_kernel<<<STAT_BLOCKS, 256, 0, stream>>>(spw, part);
    bn_finalize_kernel<<<1, 64, 0, stream>>>(part, gp, btp, scalep, shiftp);

    // 5) fused residual + ReLU + Lorentz projection -> NCHW output
    final_kernel<<<(NPIX + 255) / 256, 256, 0, stream>>>(s1, spw, scale2, shift2,
                                                         scalep, shiftp, out);
    (void)in_sizes; (void)n_in; (void)out_size; (void)ws_size;
}
